// GAT_Model_57449482551904
// MI455X (gfx1250) — matrix operand order repaired and checked by reading, not compiled, not run on hardware
//
#include <hip/hip_runtime.h>
#include <hip/hip_bf16.h>

// ---------------------------------------------------------------------------
// Types for WMMA (gfx1250, wave32)
// ---------------------------------------------------------------------------
typedef __bf16         v16bf __attribute__((ext_vector_type(16)));
typedef __bf16         v2bf  __attribute__((ext_vector_type(2)));
typedef float          v8f   __attribute__((ext_vector_type(8)));
typedef unsigned short v16u  __attribute__((ext_vector_type(16)));
typedef unsigned short u16x8 __attribute__((ext_vector_type(8)));

#define HC        256       // H*C features per node
#define NHEAD     4
#define CH        64
#define NGRAPH    64
#define OUTF      16
#define STR       40        // padded ushort stride for LDS tiles (80B, 16B aligned)
#define NEG_SLOPE 0.2f
#define EPSA      1e-16f

__device__ __forceinline__ unsigned short f2bf(float f) {
  unsigned u = __float_as_uint(f);
  u += 0x7fffu + ((u >> 16) & 1u);      // round to nearest even
  return (unsigned short)(u >> 16);
}

// pack two floats into packed bf16x2 (lo in [15:0], hi in [31:16])
__device__ __forceinline__ unsigned pack_bf16(float lo, float hi) {
#if __has_builtin(__builtin_amdgcn_cvt_pk_bf16_f32)
  v2bf p = __builtin_amdgcn_cvt_pk_bf16_f32(lo, hi);
  return __builtin_bit_cast(unsigned, p);
#else
  return (unsigned)f2bf(lo) | ((unsigned)f2bf(hi) << 16);
#endif
}

__device__ __forceinline__ void atomicMaxF(float* a, float v) {
  int* p = (int*)a;
  int old = *p;
  while (__int_as_float(old) < v) {
    int assumed = old;
    old = atomicCAS(p, assumed, __float_as_int(v));
    if (old == assumed) break;
  }
}

// ---------------------------------------------------------------------------
// GEMM: Out[M x 256] = A[M x K] @ W[K x 256], f32 in/out, bf16 WMMA compute.
// Block: 256 threads (8 waves). Block tile 64x128, wave tile 16x64 (4 WMMAs).
// ---------------------------------------------------------------------------
__global__ __launch_bounds__(256) void gemm_wmma_kernel(
    const float* __restrict__ A, const float* __restrict__ W,
    float* __restrict__ Out, int M, int K)
{
  __shared__ unsigned short sA[64 * STR];    // [row][k]   (bf16 bits)
  __shared__ unsigned short sB[128 * STR];   // [col][k]   (bf16 bits, W transposed)

  const int tid  = threadIdx.x;
  const int lane = tid & 31;
  const int wv   = tid >> 5;
  const int wm   = wv & 3;          // wave row tile (0..3)  -> rows wm*16
  const int wn   = wv >> 2;         // wave col tile (0..1)  -> cols wn*64
  const int half = lane >> 4;       // 0: lanes 0-15, 1: lanes 16-31
  const int l16  = lane & 15;
  const int brow = blockIdx.x * 64;
  const int bcol = blockIdx.y * 128;

  // staging coordinates
  const int ar = tid >> 3;          // A row     (0..31), +32 on 2nd pass
  const int ak = (tid & 7) * 4;     // A k       (0..28)
  const int bn = (tid & 31) * 4;    // B col     (0..124)
  const int bk = (tid >> 5) * 2;    // B k pair  (0..14), +16 on 2nd pass

  v8f acc0 = {}, acc1 = {}, acc2 = {}, acc3 = {};

  for (int kk = 0; kk < K; kk += 32) {
    // ---- stage A tile (64 x 32) as bf16, row-major, packed b32 stores ----
#pragma unroll
    for (int i = 0; i < 2; ++i) {
      int r  = ar + i * 32;
      int gr = brow + r;
      float4 v = make_float4(0.f, 0.f, 0.f, 0.f);
      if (gr < M) v = *(const float4*)(A + (size_t)gr * K + kk + ak);
      *(unsigned*)&sA[r * STR + ak + 0] = pack_bf16(v.x, v.y);
      *(unsigned*)&sA[r * STR + ak + 2] = pack_bf16(v.z, v.w);
    }
    // ---- stage B tile (32 x 128) transposed to [n][k]; pack adjacent k ----
#pragma unroll
    for (int i = 0; i < 2; ++i) {
      int kr = bk + i * 16;
      const float* wp = W + (size_t)(kk + kr) * HC + bcol + bn;
      float4 v0 = *(const float4*)(wp);
      float4 v1 = *(const float4*)(wp + HC);
      *(unsigned*)&sB[(bn + 0) * STR + kr] = pack_bf16(v0.x, v1.x);
      *(unsigned*)&sB[(bn + 1) * STR + kr] = pack_bf16(v0.y, v1.y);
      *(unsigned*)&sB[(bn + 2) * STR + kr] = pack_bf16(v0.z, v1.z);
      *(unsigned*)&sB[(bn + 3) * STR + kr] = pack_bf16(v0.w, v1.w);
    }
    // prefetch next K-step tiles into cache while we compute
    if (kk + 32 < K) {
      int gr = brow + ar;
      if (gr < M) __builtin_prefetch(A + (size_t)gr * K + kk + 32 + ak, 0, 3);
      __builtin_prefetch(W + (size_t)(kk + 32 + bk) * HC + bcol + bn, 0, 3);
    }
    __syncthreads();

    // A fragment: 16-bit A 16x32 layout — lane half 0 holds K {0..7,16..23},
    // half 1 holds K {8..15,24..31} (ISA 7.12.2).
    const int kb = half * 8;
    u16x8 a0 = *(const u16x8*)&sA[(wm * 16 + l16) * STR + kb];
    u16x8 a1 = *(const u16x8*)&sA[(wm * 16 + l16) * STR + kb + 16];
    v16u av;
#pragma unroll
    for (int i = 0; i < 8; ++i) { av[i] = a0[i]; av[i + 8] = a1[i]; }
    v16bf afrag = __builtin_bit_cast(v16bf, av);

    // B fragments: 16-bit B 32x16 layout — lanes 0-15 hold K {0..7,16..23},
    // lanes 16-31 hold K {8..15,24..31} (matches A fragment).
    v16bf bfrag[4];
#pragma unroll
    for (int j = 0; j < 4; ++j) {
      int col = wn * 64 + j * 16 + l16;
      u16x8 b0 = *(const u16x8*)&sB[col * STR + half * 8];
      u16x8 b1 = *(const u16x8*)&sB[col * STR + half * 8 + 16];
      v16u bv;
#pragma unroll
      for (int i = 0; i < 8; ++i) { bv[i] = b0[i]; bv[i + 8] = b1[i]; }
      bfrag[j] = __builtin_bit_cast(v16bf, bv);
    }

    acc0 = __builtin_amdgcn_wmma_f32_16x16x32_bf16(false, afrag, false, bfrag[0], (short)0, acc0, false, false);
    acc1 = __builtin_amdgcn_wmma_f32_16x16x32_bf16(false, afrag, false, bfrag[1], (short)0, acc1, false, false);
    acc2 = __builtin_amdgcn_wmma_f32_16x16x32_bf16(false, afrag, false, bfrag[2], (short)0, acc2, false, false);
    acc3 = __builtin_amdgcn_wmma_f32_16x16x32_bf16(false, afrag, false, bfrag[3], (short)0, acc3, false, false);

    __syncthreads();
  }

  // C/D layout: VGPR v -> M = v + 8*half, N = l16 (ISA 7.12.2)
#pragma unroll
  for (int v = 0; v < 8; ++v) {
    int gr = brow + wm * 16 + v + half * 8;
    if (gr < M) {
      float* o = Out + (size_t)gr * HC + bcol + wn * 64 + l16;
      o[0]  = acc0[v];
      o[16] = acc1[v];
      o[32] = acc2[v];
      o[48] = acc3[v];
    }
  }
}

// ---------------------------------------------------------------------------
// Per-node attention coefficients: a_src[n,h] = sum_c h[n,h,c]*att_src[h,c]
// One wave (32 lanes) per node; 8 channels per lane; shuffle-reduce per head.
// ---------------------------------------------------------------------------
__global__ __launch_bounds__(256) void attn_kernel(
    const float* __restrict__ h, const float* __restrict__ att_s,
    const float* __restrict__ att_d, float* __restrict__ a_src,
    float* __restrict__ a_dst, int N)
{
  int node = blockIdx.x * 8 + (threadIdx.x >> 5);
  int lane = threadIdx.x & 31;
  if (node >= N) return;
  const float* hp = h + (size_t)node * HC + lane * 8;
  float ss = 0.f, sd = 0.f;
#pragma unroll
  for (int i = 0; i < 8; ++i) {
    float v = hp[i];
    ss += v * att_s[lane * 8 + i];
    sd += v * att_d[lane * 8 + i];
  }
  // reduce across the 8 lanes that share a head (64 ch / 8 per lane)
#pragma unroll
  for (int off = 1; off < 8; off <<= 1) {
    ss += __shfl_xor(ss, off, 32);
    sd += __shfl_xor(sd, off, 32);
  }
  if ((lane & 7) == 0) {
    a_src[node * NHEAD + (lane >> 3)] = ss;
    a_dst[node * NHEAD + (lane >> 3)] = sd;
  }
}

// ---------------------------------------------------------------------------
// Init kernels
// ---------------------------------------------------------------------------
__global__ void init_md_kernel(float* __restrict__ m, float* __restrict__ den, int n)
{
  int i = blockIdx.x * blockDim.x + threadIdx.x;
  if (i < n) { m[i] = -__builtin_inff(); den[i] = 0.f; }
}

__global__ void zero_kernel(float* __restrict__ p, int n)
{
  int i = blockIdx.x * blockDim.x + threadIdx.x;
  if (i < n) p[i] = 0.f;
}

__global__ void init_readout_kernel(float* __restrict__ gmx, float* __restrict__ gsm,
                                    float* __restrict__ cnt)
{
  int i = blockIdx.x * blockDim.x + threadIdx.x;
  if (i < NGRAPH * CH) { gmx[i] = -__builtin_inff(); gsm[i] = 0.f; }
  if (i < NGRAPH) cnt[i] = 0.f;
}

// ---------------------------------------------------------------------------
// Edge pass 1: e = leaky_relu(a_src[src]+a_dst[dst]); segment max into m[dst]
// Edges: [0,E) from edge_index, [E,E+N) are self loops.
// ---------------------------------------------------------------------------
__global__ void edge_max_kernel(const int* __restrict__ ei,
                                const float* __restrict__ a_s,
                                const float* __restrict__ a_d,
                                float* __restrict__ ev, float* __restrict__ m,
                                int E, int ET4)
{
  int i = blockIdx.x * blockDim.x + threadIdx.x;
  if (i >= ET4) return;
  int e = i >> 2, hh = i & 3;
  int s, d;
  if (e < E) { s = ei[e]; d = ei[E + e]; } else { s = d = e - E; }
  float x = a_s[s * NHEAD + hh] + a_d[d * NHEAD + hh];
  float lr = (x > 0.f) ? x : NEG_SLOPE * x;
  ev[i] = lr;
  atomicMaxF(&m[d * NHEAD + hh], lr);
}

// Edge pass 2: denom[dst] += exp(e - m[dst])
__global__ void edge_sum_kernel(const int* __restrict__ ei,
                                const float* __restrict__ ev,
                                const float* __restrict__ m,
                                float* __restrict__ den, int E, int ET4)
{
  int i = blockIdx.x * blockDim.x + threadIdx.x;
  if (i >= ET4) return;
  int e = i >> 2, hh = i & 3;
  int d = (e < E) ? ei[E + e] : (e - E);
  float ex = expf(ev[i] - m[d * NHEAD + hh]);
  atomicAdd(&den[d * NHEAD + hh], ex);
}

// Edge pass 3: agg[dst, :] += h[src, :] * alpha  (256 threads = 1 edge)
__global__ __launch_bounds__(256) void edge_msg_kernel(
    const int* __restrict__ ei, const float* __restrict__ h,
    const float* __restrict__ ev, const float* __restrict__ m,
    const float* __restrict__ den, float* __restrict__ agg, int E)
{
  int e   = blockIdx.x;
  int tid = threadIdx.x;
  int s, d;
  if (e < E) { s = ei[e]; d = ei[E + e]; } else { s = d = e - E; }
  int hh = tid >> 6;
  float alpha = expf(ev[e * NHEAD + hh] - m[d * NHEAD + hh]) /
                (den[d * NHEAD + hh] + EPSA);
  atomicAdd(&agg[(size_t)d * HC + tid], h[(size_t)s * HC + tid] * alpha);
}

// bias + tanh, in place
__global__ void bias_tanh_kernel(float* __restrict__ p, const float* __restrict__ b, int n)
{
  int i = blockIdx.x * blockDim.x + threadIdx.x;
  if (i < n) p[i] = tanhf(p[i] + b[i & (HC - 1)]);
}

// ---------------------------------------------------------------------------
// Readout: head-mean then per-graph segment max / sum / count
// 64 threads per node (4 nodes per 256-thread block)
// ---------------------------------------------------------------------------
__global__ __launch_bounds__(256) void readout_kernel(
    const float* __restrict__ h, const int* __restrict__ batch,
    float* __restrict__ gmx, float* __restrict__ gsm, float* __restrict__ cnt, int N)
{
  int node = blockIdx.x * 4 + (threadIdx.x >> 6);
  int c    = threadIdx.x & 63;
  if (node >= N) return;
  const float* p = h + (size_t)node * HC;
  float v = 0.25f * (p[c] + p[CH + c] + p[2 * CH + c] + p[3 * CH + c]);
  int b = batch[node];
  atomicMaxF(&gmx[b * CH + c], v);
  atomicAdd(&gsm[b * CH + c], v);
  if (c == 0) atomicAdd(&cnt[b], 1.0f);
}

// Final: hidden = [gmax | gmean | gsum]; out = hidden @ Wout + bout
// d_out layout: out[64*16] then hidden[64*192]
__global__ __launch_bounds__(192) void final_kernel(
    const float* __restrict__ gmx, const float* __restrict__ gsm,
    const float* __restrict__ cnt, const float* __restrict__ Wout,
    const float* __restrict__ bout, float* __restrict__ out)
{
  __shared__ float hid[3 * CH];
  int b = blockIdx.x, j = threadIdx.x;   // j in [0,192)
  float v;
  if (j < CH)            v = gmx[b * CH + j];
  else if (j < 2 * CH)   v = gsm[b * CH + (j - CH)] / fmaxf(cnt[b], 1.f);
  else                   v = gsm[b * CH + (j - 2 * CH)];
  hid[j] = v;
  out[NGRAPH * OUTF + b * 3 * CH + j] = v;     // hidden output
  __syncthreads();
  if (j < OUTF) {
    float acc = bout[j];
#pragma unroll 4
    for (int k = 0; k < 3 * CH; ++k) acc += hid[k] * Wout[k * OUTF + j];
    out[b * OUTF + j] = acc;
  }
}

// ---------------------------------------------------------------------------
// Launcher
// ---------------------------------------------------------------------------
extern "C" void kernel_launch(void* const* d_in, const int* in_sizes, int n_in,
                              void* d_out, int out_size, void* d_ws, size_t ws_size,
                              hipStream_t stream)
{
  const float* x    = (const float*)d_in[0];
  const int*   ei   = (const int*)d_in[1];
  const int*   bi   = (const int*)d_in[2];
  const float* Wl[3]  = { (const float*)d_in[3], (const float*)d_in[7],  (const float*)d_in[11] };
  const float* As[3]  = { (const float*)d_in[4], (const float*)d_in[8],  (const float*)d_in[12] };
  const float* Ad[3]  = { (const float*)d_in[5], (const float*)d_in[9],  (const float*)d_in[13] };
  const float* Bl[3]  = { (const float*)d_in[6], (const float*)d_in[10], (const float*)d_in[14] };
  const float* Wout = (const float*)d_in[15];
  const float* bout = (const float*)d_in[16];

  const int N  = in_sizes[2];        // batch_index length
  const int E  = in_sizes[1] / 2;    // edge_index is [2, E]
  const int ET = E + N;              // with self loops
  const int ET4 = ET * NHEAD;

  // workspace layout (floats)
  float* base = (float*)d_ws;
  float* hA  = base;                       // N*256: transformed features
  float* hB  = hA  + (size_t)N * HC;       // N*256: aggregate / layer input
  float* aS  = hB  + (size_t)N * HC;       // N*4
  float* aD  = aS  + (size_t)N * NHEAD;    // N*4
  float* mM  = aD  + (size_t)N * NHEAD;    // N*4
  float* den = mM  + (size_t)N * NHEAD;    // N*4
  float* ev  = den + (size_t)N * NHEAD;    // ET*4
  float* gmx = ev  + (size_t)ET * NHEAD;   // 64*64
  float* gsm = gmx + NGRAPH * CH;          // 64*64
  float* cnt = gsm + NGRAPH * CH;          // 64

  dim3 blk(256);
  int edgeGrid = (ET4 + 255) / 256;

  init_readout_kernel<<<(NGRAPH * CH + 255) / 256, blk, 0, stream>>>(gmx, gsm, cnt);

  const float* in = x;
  for (int l = 0; l < 3; ++l) {
    int K = (l == 0) ? 128 : HC;

    dim3 ggrid((N + 63) / 64, HC / 128);
    gemm_wmma_kernel<<<ggrid, blk, 0, stream>>>(in, Wl[l], hA, N, K);

    attn_kernel<<<(N + 7) / 8, blk, 0, stream>>>(hA, As[l], Ad[l], aS, aD, N);
    init_md_kernel<<<(N * NHEAD + 255) / 256, blk, 0, stream>>>(mM, den, N * NHEAD);
    zero_kernel<<<((int)((size_t)N * HC) + 255) / 256, blk, 0, stream>>>(hB, N * HC);

    edge_max_kernel<<<edgeGrid, blk, 0, stream>>>(ei, aS, aD, ev, mM, E, ET4);
    edge_sum_kernel<<<edgeGrid, blk, 0, stream>>>(ei, ev, mM, den, E, ET4);
    edge_msg_kernel<<<ET, blk, 0, stream>>>(ei, hA, ev, mM, den, hB, E);

    bias_tanh_kernel<<<((int)((size_t)N * HC) + 255) / 256, blk, 0, stream>>>(hB, Bl[l], N * HC);
    in = hB;
  }

  readout_kernel<<<(N + 3) / 4, blk, 0, stream>>>(hB, bi, gmx, gsm, cnt, N);
  final_kernel<<<NGRAPH, dim3(3 * CH), 0, stream>>>(gmx, gsm, cnt, Wout, bout, (float*)d_out);
}
